// FungalBrain_15195594293753
// MI455X (gfx1250) — matrix operand order repaired and checked by reading, not compiled, not run on hardware
//
#include <hip/hip_runtime.h>

// ---------------------------------------------------------------------------
// FungalBrain on MI455X (gfx1250, wave32, WMMA bf16 16x16x32, fp32 accum)
// ---------------------------------------------------------------------------

#define FB_T     2048
#define FB_B     32

// output offsets (floats)
#define O_PRED 0
#define O_HID  1048064u
#define O_ANOM 34602496u
#define O_SURP 34668032u
#define O_PROP 34733536u
#define O_NODE 34733600u
#define O_POOL 61472288u

// workspace offsets (bytes)
#define WS_WD    0u          // 8*4*2*512  bf16
#define WS_NW    65536u      // 8*4*8*512  bf16
#define WS_PW    327680u     // 4*15*512   bf16
#define WS_WT    389120u     // 32*15*512  bf16
#define WS_WTP   880640u     // 32*15*512  bf16
#define WS_DIFF  1372160u    // 64 f32
#define WS_NSTEP 1372416u    // 408 f32
#define WS_PSTEP 1374048u    // 51 f32
#define WS_ZERO  1374252u    // 32 f32 of zeros (async zero-pad source)

typedef __attribute__((ext_vector_type(16))) __bf16 v16bf;
typedef __attribute__((ext_vector_type(8)))  float  v8f;

union V16U { v16bf v; unsigned short u[16]; };

struct FBParams { const float* p[41]; };

// hardware f32 -> bf16 (RNE) via __bf16 cast; stored as raw ushort
__device__ __forceinline__ unsigned short f2bf(float f) {
    union { __bf16 b; unsigned short u; } v;
    v.b = (__bf16)f;
    return v.u;
}

// A operand: row-major bf16 (ushort) LDS, rowPtr points at (M, K-chunk base).
// lane L: M = L&15, half = L>>4; elems 0..15 -> K = half*16+e
__device__ __forceinline__ v16bf loadA(const unsigned short* rowPtr, int half) {
    V16U a;
    const unsigned short* p0 = rowPtr + half * 16;
    const unsigned short* p1 = rowPtr + half * 16 + 8;
#pragma unroll
    for (int e = 0; e < 8; ++e) { a.u[e] = p0[e]; a.u[8 + e] = p1[e]; }
    return a.v;
}

// A operand assembled from fp32 LDS (async-staged) with hardware cvt at read.
__device__ __forceinline__ v16bf loadA_f32(const float* rowPtr, int half) {
    V16U a;
    const float* p0 = rowPtr + half * 16;
    const float* p1 = rowPtr + half * 16 + 8;
#pragma unroll
    for (int e = 0; e < 8; ++e) { a.u[e] = f2bf(p0[e]); a.u[8 + e] = f2bf(p1[e]); }
    return a.v;
}

// B operand: pre-packed in exact lane order (tile*512 + lane*16 + e), contiguous per lane.
__device__ __forceinline__ v16bf loadB(const unsigned short* g) {
    V16U b;
#pragma unroll
    for (int e = 0; e < 16; ++e) b.u[e] = g[e];
    return b.v;
}

__device__ __forceinline__ v8f wmma_bf16(v16bf a, v16bf b, v8f c) {
    return __builtin_amdgcn_wmma_f32_16x16x32_bf16(false, a, false, b, (short)0, c,
                                                   false, false);
}

// async copy one dword global -> LDS (ASYNCcnt-tracked)
__device__ __forceinline__ void async_g2l_b32(float* ldsDst, const float* gsrc) {
    unsigned ldsoff = (unsigned)(unsigned long long)ldsDst;     // low 32 bits = LDS offset
    asm volatile("global_load_async_to_lds_b32 %0, %1, off"
                 :: "v"(ldsoff), "v"((unsigned long long)gsrc) : "memory");
}

// ---------------------------------------------------------------------------
// prep: softmax(diff), step sizes, async zero-pad block
// ---------------------------------------------------------------------------
__global__ void fb_prep_small(FBParams P, unsigned char* ws) {
    const int tid = threadIdx.x;
    float* diffw = (float*)(ws + WS_DIFF);
    float* nstep = (float*)(ws + WS_NSTEP);
    float* pstep = (float*)(ws + WS_PSTEP);
    float* zblk  = (float*)(ws + WS_ZERO);
    const float* chem = P.p[3];
    const float* nls  = P.p[10];
    const float* pls  = P.p[17];
    if (tid < 8) {
        float mx = -1e30f;
#pragma unroll
        for (int j = 0; j < 8; ++j) mx = fmaxf(mx, chem[tid * 8 + j]);
        float e[8]; float s = 0.f;
#pragma unroll
        for (int j = 0; j < 8; ++j) { e[j] = expf(chem[tid * 8 + j] - mx); s += e[j]; }
        float inv = 1.f / s;
#pragma unroll
        for (int j = 0; j < 8; ++j) diffw[tid * 8 + j] = e[j] * inv;
    } else if (tid >= 16 && tid < 424) {
        nstep[tid - 16] = 0.3f * expf(nls[tid - 16]);        // dt * dt_scale = 0.3
    } else if (tid >= 424 && tid < 475) {
        pstep[tid - 424] = 0.3f * expf(pls[tid - 424]);
    } else if (tid >= 480) {
        zblk[tid - 480] = 0.f;
    }
}

// ---------------------------------------------------------------------------
// prep: pack all B matrices into WMMA lane order, bf16, zero-padded
// ---------------------------------------------------------------------------
__global__ void fb_prep_pack(FBParams P, unsigned char* ws) {
    const float* Wd  = P.p[4];
    const float* nWi = P.p[6];
    const float* nWh = P.p[8];
    const float* pWi = P.p[13];
    const float* pWh = P.p[15];
    const float* Wt  = P.p[20];
    const float* Wtp = P.p[24];
    unsigned short* wd_pk  = (unsigned short*)(ws + WS_WD);
    unsigned short* nw_pk  = (unsigned short*)(ws + WS_NW);
    unsigned short* pw_pk  = (unsigned short*)(ws + WS_PW);
    unsigned short* wt_pk  = (unsigned short*)(ws + WS_WT);
    unsigned short* wtp_pk = (unsigned short*)(ws + WS_WTP);
    const int nwd = 32768, nnw = 131072, npw = 30720, nwt = 245760;
    const int total = nwd + nnw + npw + nwt + nwt;
    for (int idx = blockIdx.x * blockDim.x + threadIdx.x; idx < total;
         idx += gridDim.x * blockDim.x) {
        float val = 0.f;
        if (idx < nwd) {
            int u = idx, tile = u >> 9, r = u & 511, lane = r >> 4, e = r & 15;
            int kc = tile & 1, nt = (tile >> 1) & 3, j = tile >> 3;
            int K = kc * 32 + (lane >> 4) * 16 + e;
            int N = nt * 16 + (lane & 15);
            if (K < 51 && N < 51) val = Wd[(j * 51 + K) * 51 + N];
            wd_pk[u] = f2bf(val);
        } else if (idx < nwd + nnw) {
            int u = idx - nwd, tile = u >> 9, r = u & 511, lane = r >> 4, e = r & 15;
            int kc = tile & 7, nt = (tile >> 3) & 3, i = tile >> 5;
            int K = kc * 32 + (lane >> 4) * 16 + e;
            int N = nt * 16 + (lane & 15);
            if (N < 51) {
                if (K < 204)      val = nWi[(i * 204 + K) * 51 + N];
                else if (K < 255) val = nWh[(i * 51 + (K - 204)) * 51 + N];
            }
            nw_pk[u] = f2bf(val);
        } else if (idx < nwd + nnw + npw) {
            int u = idx - (nwd + nnw), tile = u >> 9, r = u & 511, lane = r >> 4, e = r & 15;
            int kc = tile % 15, nt = tile / 15;
            int K = kc * 32 + (lane >> 4) * 16 + e;
            int N = nt * 16 + (lane & 15);
            if (N < 51) {
                if (K < 408)      val = pWi[K * 51 + N];
                else if (K < 459) val = pWh[(K - 408) * 51 + N];
            }
            pw_pk[u] = f2bf(val);
        } else if (idx < nwd + nnw + npw + nwt) {
            int u = idx - (nwd + nnw + npw), tile = u >> 9, r = u & 511, lane = r >> 4, e = r & 15;
            int kc = tile % 15, nt = tile / 15;
            int K = kc * 32 + (lane >> 4) * 16 + e;
            int N = nt * 16 + (lane & 15);
            if (K < 459) val = Wt[K * 512 + N];
            wt_pk[u] = f2bf(val);
        } else {
            int u = idx - (nwd + nnw + npw + nwt), tile = u >> 9, r = u & 511, lane = r >> 4, e = r & 15;
            int kc = tile % 15, nt = tile / 15;
            int K = kc * 32 + (lane >> 4) * 16 + e;
            int N = nt * 16 + (lane & 15);
            if (K < 459) val = Wtp[K * 512 + N];
            wtp_pk[u] = f2bf(val);
        }
    }
}

// ---------------------------------------------------------------------------
// Phase 1: the recurrence. 2 workgroups x 16 batch rows, 16 waves, all state
// in LDS, WMMA for chem / node-cell / pool GEMMs each step.
// ---------------------------------------------------------------------------
__global__ __launch_bounds__(512)
void fb_recurrence(FBParams P, const unsigned char* ws, float* out) {
    extern __shared__ unsigned char smem[];
    float* nodes = (float*)smem;                     // [16][8][64] fp32 (pad cols zero)
    float* pool  = nodes + 16 * 8 * 64;              // [16][64]
    float* inpb  = pool + 16 * 64;                   // [16][64]   (reused as pool-pre)
    float* ybuf  = inpb + 16 * 64;                   // [16][8][64](reused as node-pre)
    unsigned short* nodesA = (unsigned short*)(ybuf + 16 * 8 * 64); // [8][16][64]
    unsigned short* cellA  = nodesA + 8 * 16 * 64;   // [8][16][256]
    unsigned short* poolA  = cellA + 8 * 16 * 256;   // [16][480]
    float* diffs = (float*)(poolA + 16 * 480);       // [64]

    const int tid = threadIdx.x;
    const int wave = tid >> 5, lane = tid & 31;
    const int half = lane >> 4, mrow = lane & 15;
    const int wg = blockIdx.x;

    const unsigned short* wd_pk = (const unsigned short*)(ws + WS_WD);
    const unsigned short* nw_pk = (const unsigned short*)(ws + WS_NW);
    const unsigned short* pw_pk = (const unsigned short*)(ws + WS_PW);
    const float* diffw = (const float*)(ws + WS_DIFF);
    const float* nstep = (const float*)(ws + WS_NSTEP);
    const float* pstep = (const float*)(ws + WS_PSTEP);

    const float* x      = P.p[0];
    const float* Wi_in  = P.p[1];
    const float* bi_in  = P.p[2];
    const float* bd     = P.p[5];
    const float* nbi    = P.p[7];
    const float* nbh    = P.p[9];
    const float* nlg    = P.p[11];
    const float* nlb    = P.p[12];
    const float* pbi    = P.p[14];
    const float* pbh    = P.p[16];
    const float* plg    = P.p[18];
    const float* plb    = P.p[19];

    for (int i = tid; i < 16 * 8 * 64; i += 512) nodes[i] = 0.f;
    for (int i = tid; i < 16 * 64; i += 512) pool[i] = 0.f;
    if (tid < 64) diffs[tid] = diffw[tid];
    __syncthreads();

    for (int t = 0; t < FB_T; ++t) {
        // S1: input proj (VALU) + stage nodes -> bf16 A for chem stage
        for (int idx = tid; idx < 16 * 64; idx += 512) {
            int mb = idx >> 6, c = idx & 63;
            float v = 0.f;
            if (c < 51) {
                v = bi_in[c];
                const float* xr = x + ((size_t)(wg * 16 + mb) * FB_T + t) * 16;
#pragma unroll
                for (int i2 = 0; i2 < 16; ++i2) v += xr[i2] * Wi_in[i2 * 51 + c];
            }
            inpb[idx] = v;
        }
        for (int idx = tid; idx < 8 * 16 * 64; idx += 512) {
            int j = idx >> 10; int rem = idx & 1023; int mb = rem >> 6; int c = rem & 63;
            nodesA[idx] = f2bf(nodes[(mb * 8 + j) * 64 + c]);
        }
        __syncthreads();

        // S2: y[j] = nodes_j @ Wd_j   (8 nodes x 4 ntiles, K=64 -> 2 chunks)
        for (int task = wave; task < 32; task += 16) {
            int j = task >> 2, nt = task & 3;
            v8f acc = {};
#pragma unroll
            for (int kc = 0; kc < 2; ++kc) {
                v16bf a = loadA(nodesA + (j * 16 + mrow) * 64 + kc * 32, half);
                v16bf b = loadB(wd_pk + (size_t)((j * 4 + nt) * 2 + kc) * 512 + lane * 16);
                acc = wmma_bf16(a, b, acc);
            }
#pragma unroll
            for (int r = 0; r < 8; ++r) {
                int row = r + 8 * half; int col = nt * 16 + mrow;
                ybuf[(row * 8 + j) * 64 + col] = acc[r];
            }
        }
        __syncthreads();

        // S3: chem mix + build cell A [inp|chem|pool|nodes|nodes] (K=256), pool A tail
        for (int idx = tid; idx < 8 * 16 * 256; idx += 512) {
            int i = idx >> 12; int rem = idx & 4095; int mb = rem >> 8; int k = rem & 255;
            float v;
            if (k < 51) v = inpb[mb * 64 + k];
            else if (k < 102) {
                int c = k - 51; float s = bd[c];
#pragma unroll
                for (int j = 0; j < 8; ++j) s += diffs[i * 8 + j] * ybuf[(mb * 8 + j) * 64 + c];
                v = s;
            } else if (k < 153) v = pool[mb * 64 + (k - 102)];
            else if (k < 204)   v = nodes[(mb * 8 + i) * 64 + (k - 153)];
            else if (k < 255)   v = nodes[(mb * 8 + i) * 64 + (k - 204)];   // nWh fold
            else v = 0.f;
            cellA[idx] = f2bf(v);
        }
        for (int idx = tid; idx < 16 * 72; idx += 512) {                    // old pool -> pWh fold
            int mb = idx / 72, c = idx % 72; int col = 408 + c;
            float v = (col < 459) ? pool[mb * 64 + (col - 408)] : 0.f;
            poolA[mb * 480 + col] = f2bf(v);
        }
        __syncthreads();

        // S4: node pre = cell @ nW ; Euler + tanh update (pre-LN)
        for (int task = wave; task < 32; task += 16) {
            int i = task >> 2, nt = task & 3;
            v8f acc = {};
#pragma unroll
            for (int kc = 0; kc < 8; ++kc) {
                v16bf a = loadA(cellA + (i * 16 + mrow) * 256 + kc * 32, half);
                v16bf b = loadB(nw_pk + (size_t)((i * 4 + nt) * 8 + kc) * 512 + lane * 16);
                acc = wmma_bf16(a, b, acc);
            }
#pragma unroll
            for (int r = 0; r < 8; ++r) {
                int row = r + 8 * half; int col = nt * 16 + mrow;
                if (col < 51) {
                    float pre = acc[r] + nbi[i * 51 + col] + nbh[i * 51 + col];
                    float nold = nodes[(row * 8 + i) * 64 + col];
                    ybuf[(row * 8 + i) * 64 + col] =
                        nold + nstep[i * 51 + col] * (tanhf(pre) - nold);
                }
            }
        }
        __syncthreads();

        // S5: per-node LN, write node_tr, stage poolA node part
        if (tid < 128) {
            int mb = tid >> 3, i = tid & 7;
            const float* pr = ybuf + (mb * 8 + i) * 64;
            float m = 0.f;
            for (int h = 0; h < 51; ++h) m += pr[h];
            m *= (1.f / 51.f);
            float v = 0.f;
            for (int h = 0; h < 51; ++h) { float d = pr[h] - m; v += d * d; }
            float rstd = rsqrtf(v * (1.f / 51.f) + 1e-5f);
            size_t bt = (size_t)(wg * 16 + mb) * FB_T + t;
            float* ndst = out + O_NODE + bt * 408 + i * 51;
            for (int h = 0; h < 51; ++h) {
                float val = (pr[h] - m) * rstd * nlg[i * 51 + h] + nlb[i * 51 + h];
                nodes[(mb * 8 + i) * 64 + h] = val;
                ndst[h] = val;
                poolA[mb * 480 + i * 51 + h] = f2bf(val);
            }
        }
        __syncthreads();

        // S6: pool pre = [pool_in|pool] @ pW ; update (pre-LN)
        if (wave < 4) {
            int nt = wave; v8f acc = {};
#pragma unroll
            for (int kc = 0; kc < 15; ++kc) {
                v16bf a = loadA(poolA + mrow * 480 + kc * 32, half);
                v16bf b = loadB(pw_pk + (size_t)(nt * 15 + kc) * 512 + lane * 16);
                acc = wmma_bf16(a, b, acc);
            }
#pragma unroll
            for (int r = 0; r < 8; ++r) {
                int row = r + 8 * half; int col = nt * 16 + mrow;
                if (col < 51) {
                    float pre = acc[r] + pbi[col] + pbh[col];
                    float pold = pool[row * 64 + col];
                    inpb[row * 64 + col] = pold + pstep[col] * (tanhf(pre) - pold);
                }
            }
        }
        __syncthreads();

        // S7: pool LN, write pool_tr
        if (tid < 16) {
            int mb = tid;
            const float* pr = inpb + mb * 64;
            float m = 0.f;
            for (int h = 0; h < 51; ++h) m += pr[h];
            m *= (1.f / 51.f);
            float v = 0.f;
            for (int h = 0; h < 51; ++h) { float d = pr[h] - m; v += d * d; }
            float rstd = rsqrtf(v * (1.f / 51.f) + 1e-5f);
            size_t bt = (size_t)(wg * 16 + mb) * FB_T + t;
            float* pdst = out + O_POOL + bt * 51;
            for (int h = 0; h < 51; ++h) {
                float val = (pr[h] - m) * rstd * plg[h] + plb[h];
                pool[mb * 64 + h] = val;
                pdst[h] = val;
            }
        }
        __syncthreads();
    }
}

// ---------------------------------------------------------------------------
// Phase 2: hidden[bt] = sigmoid(LN(allr@Wt+bt)) * (allr@Wtp+btp)
// BM=32 rows/block, BN=512 (full, for row LN), 16 waves x 8 WMMA per K chunk.
// A tile is double-buffered and staged with GLOBAL_LOAD_ASYNC_TO_LDS_B32
// (ASYNCcnt), overlapping global traffic with the matrix pipe.
// ---------------------------------------------------------------------------
__global__ __launch_bounds__(512)
void fb_thalamus(FBParams P, const unsigned char* ws, float* out) {
    extern __shared__ unsigned char smem[];
    float* Abuf  = (float*)smem;                     // [2][32][32] fp32 (double buffer)
    float* hidg  = (float*)(smem + 8192);            // [32][512]
    float* hidt  = hidg + 32 * 512;                  // [32][512]
    float* red   = hidt + 32 * 512;                  // [512]
    float* red2  = red + 512;                        // [512]
    float* stats = red2 + 512;                       // [64]

    const int tid = threadIdx.x;
    const int wave = tid >> 5, lane = tid & 31;
    const int half = lane >> 4, mrow = lane & 15;
    const int bt0 = blockIdx.x * 32;

    const unsigned short* wt_pk  = (const unsigned short*)(ws + WS_WT);
    const unsigned short* wtp_pk = (const unsigned short*)(ws + WS_WTP);
    const float* zblk = (const float*)(ws + WS_ZERO);
    const float* btl  = P.p[21];
    const float* tlg  = P.p[22];
    const float* tlb  = P.p[23];
    const float* btpl = P.p[25];

    v8f z = {};
    v8f ag00 = z, ag01 = z, ag10 = z, ag11 = z;
    v8f at00 = z, at01 = z, at10 = z, at11 = z;
    const int nt0 = wave * 2, nt1 = wave * 2 + 1;

    // issue exactly 2 async dword copies per thread for chunk kc -> Abuf[kc&1]
    auto issueA = [&](int kc) {
        float* buf = Abuf + (kc & 1) * 1024;
#pragma unroll
        for (int q = 0; q < 2; ++q) {
            int idx = tid * 2 + q;
            int row = idx >> 5, k = idx & 31;
            int K = kc * 32 + k; int bt = bt0 + row;
            const float* gp;
            if (K < 408)      gp = out + O_NODE + (size_t)bt * 408 + K;
            else if (K < 459) gp = out + O_POOL + (size_t)bt * 51 + (K - 408);
            else              gp = zblk + (K - 459);          // zero padding source
            async_g2l_b32(buf + idx, gp);
        }
    };

    issueA(0);
    for (int kc = 0; kc < 15; ++kc) {
        if (kc < 14) {
            issueA(kc + 1);                                   // overlap next copy
            __builtin_prefetch(wt_pk + (size_t)(nt0 * 15 + kc + 1) * 512 + lane * 16, 0, 1);
            __builtin_prefetch(wtp_pk + (size_t)(nt0 * 15 + kc + 1) * 512 + lane * 16, 0, 1);
            asm volatile("s_wait_asynccnt 0x2" ::: "memory"); // drain current 2 (in-order)
        } else {
            asm volatile("s_wait_asynccnt 0x0" ::: "memory");
        }
        __syncthreads();                                      // whole tile visible

        const float* At = Abuf + (kc & 1) * 1024;
        v16bf a0 = loadA_f32(At + mrow * 32, half);
        v16bf a1 = loadA_f32(At + (16 + mrow) * 32, half);
        v16bf bg0 = loadB(wt_pk + (size_t)(nt0 * 15 + kc) * 512 + lane * 16);
        v16bf bg1 = loadB(wt_pk + (size_t)(nt1 * 15 + kc) * 512 + lane * 16);
        v16bf bp0 = loadB(wtp_pk + (size_t)(nt0 * 15 + kc) * 512 + lane * 16);
        v16bf bp1 = loadB(wtp_pk + (size_t)(nt1 * 15 + kc) * 512 + lane * 16);
        ag00 = wmma_bf16(a0, bg0, ag00); ag01 = wmma_bf16(a0, bg1, ag01);
        ag10 = wmma_bf16(a1, bg0, ag10); ag11 = wmma_bf16(a1, bg1, ag11);
        at00 = wmma_bf16(a0, bp0, at00); at01 = wmma_bf16(a0, bp1, at01);
        at10 = wmma_bf16(a1, bp0, at10); at11 = wmma_bf16(a1, bp1, at11);
        __syncthreads();                                      // done reading this buffer
    }

    auto store_t = [&](v8f a, float* dst, const float* bias, int row0, int col0) {
#pragma unroll
        for (int r = 0; r < 8; ++r) {
            int row = row0 + r + 8 * half;
            int col = col0 + mrow;
            dst[row * 512 + col] = a[r] + bias[col];
        }
    };
    store_t(ag00, hidg, btl, 0,  nt0 * 16); store_t(ag01, hidg, btl, 0,  nt1 * 16);
    store_t(ag10, hidg, btl, 16, nt0 * 16); store_t(ag11, hidg, btl, 16, nt1 * 16);
    store_t(at00, hidt, btpl, 0,  nt0 * 16); store_t(at01, hidt, btpl, 0,  nt1 * 16);
    store_t(at10, hidt, btpl, 16, nt0 * 16); store_t(at11, hidt, btpl, 16, nt1 * 16);
    __syncthreads();

    // row LN over 512 (16 threads per row)
    {
        int row = tid >> 4, part = tid & 15;
        float s = 0.f, s2 = 0.f;
        for (int c = part * 32; c < part * 32 + 32; ++c) {
            float g = hidg[row * 512 + c]; s += g; s2 += g * g;
        }
        red[tid] = s; red2[tid] = s2;
        __syncthreads();
        if (part == 0) {
            float S = 0.f, S2 = 0.f;
            for (int p2 = 0; p2 < 16; ++p2) { S += red[row * 16 + p2]; S2 += red2[row * 16 + p2]; }
            float mean = S * (1.f / 512.f);
            float var = S2 * (1.f / 512.f) - mean * mean;
            stats[row * 2] = mean;
            stats[row * 2 + 1] = rsqrtf(var + 1e-5f);
        }
        __syncthreads();
    }
    for (int idx = tid; idx < 32 * 512; idx += 512) {
        int row = idx >> 9, col = idx & 511;
        float mean = stats[row * 2], rstd = stats[row * 2 + 1];
        float ln = (hidg[idx] - mean) * rstd * tlg[col] + tlb[col];
        float gate = 1.f / (1.f + expf(-ln));
        out[O_HID + (size_t)(bt0 + row) * 512 + col] = gate * hidt[idx];
    }
}

// ---------------------------------------------------------------------------
// Phase 3: predictions / anomaly / surprise (wave-per-row)
// ---------------------------------------------------------------------------
__global__ __launch_bounds__(256)
void fb_heads(FBParams P, float* out) {
    const int tid = threadIdx.x;
    const int lane = tid & 31;
    const int bt = blockIdx.x * 8 + (tid >> 5);
    const float* x   = P.p[0];
    const float* nsg = P.p[26]; const float* nsb = P.p[27];
    const float* Wns = P.p[28]; const float* bns = P.p[29];
    const float* ang = P.p[30]; const float* anb = P.p[31];
    const float* Wan = P.p[32]; const float* ban = P.p[33];

    const float* hrow = out + O_HID + (size_t)bt * 512;
    float h[16];
#pragma unroll
    for (int i = 0; i < 16; ++i) h[i] = hrow[lane + 32 * i];
    float s = 0.f, s2 = 0.f;
#pragma unroll
    for (int i = 0; i < 16; ++i) { s += h[i]; s2 += h[i] * h[i]; }
#pragma unroll
    for (int m = 16; m >= 1; m >>= 1) { s += __shfl_xor(s, m, 32); s2 += __shfl_xor(s2, m, 32); }
    float mean = s * (1.f / 512.f);
    float var = s2 * (1.f / 512.f) - mean * mean;
    float rstd = rsqrtf(var + 1e-5f);
    float core[16];
#pragma unroll
    for (int i = 0; i < 16; ++i) core[i] = (h[i] - mean) * rstd;

    float pa = 0.f;
#pragma unroll
    for (int i = 0; i < 16; ++i) { int k = lane + 32 * i; pa += (core[i] * ang[k] + anb[k]) * Wan[k]; }
#pragma unroll
    for (int m = 16; m >= 1; m >>= 1) pa += __shfl_xor(pa, m, 32);
    if (lane == 0) out[O_ANOM + bt] = 1.f / (1.f + expf(-(pa + ban[0])));

    const int b = bt >> 11, t = bt & 2047;
    if (t < 2047) {
        float lnv[16];
#pragma unroll
        for (int i = 0; i < 16; ++i) { int k = lane + 32 * i; lnv[i] = core[i] * nsg[k] + nsb[k]; }
        float serr = 0.f;
#pragma unroll
        for (int j = 0; j < 16; ++j) {
            float pj = 0.f;
#pragma unroll
            for (int i = 0; i < 16; ++i) { int k = lane + 32 * i; pj += lnv[i] * Wns[k * 16 + j]; }
#pragma unroll
            for (int m = 16; m >= 1; m >>= 1) pj += __shfl_xor(pj, m, 32);
            if (lane == j) {
                float pv = pj + bns[j];
                out[O_PRED + ((size_t)b * 2047 + t) * 16 + j] = pv;
                float d = pv - x[((size_t)b * 2048 + (t + 1)) * 16 + j];
                serr = d * d;
            }
        }
#pragma unroll
        for (int m = 16; m >= 1; m >>= 1) serr += __shfl_xor(serr, m, 32);
        if (lane == 0) out[O_SURP + (size_t)b * 2047 + t] = serr * (1.f / 16.f);
    }
}

// ---------------------------------------------------------------------------
// Phase 4: properties head (single block)
// ---------------------------------------------------------------------------
__global__ __launch_bounds__(256)
void fb_props(FBParams P, float* out) {
    __shared__ float lnh[512];
    __shared__ float h1[256];
    __shared__ float red[256];
    __shared__ float stat[2];
    const int tid = threadIdx.x;
    const float* prg = P.p[34]; const float* prb = P.p[35];
    const float* Wp1 = P.p[36]; const float* bp1 = P.p[37];
    const float* Wp2 = P.p[38]; const float* bp2 = P.p[39];

    for (int b = 0; b < 32; ++b) {
        const float* hrow = out + O_HID + ((size_t)b * 2048 + 2047) * 512;
        float v0 = hrow[2 * tid], v1 = hrow[2 * tid + 1];
        red[tid] = v0 + v1;
        __syncthreads();
        for (int off = 128; off; off >>= 1) { if (tid < off) red[tid] += red[tid + off]; __syncthreads(); }
        if (tid == 0) stat[0] = red[0] * (1.f / 512.f);
        __syncthreads();
        float mean = stat[0];
        red[tid] = (v0 - mean) * (v0 - mean) + (v1 - mean) * (v1 - mean);
        __syncthreads();
        for (int off = 128; off; off >>= 1) { if (tid < off) red[tid] += red[tid + off]; __syncthreads(); }
        if (tid == 0) stat[1] = rsqrtf(red[0] * (1.f / 512.f) + 1e-5f);
        __syncthreads();
        float rstd = stat[1];
        lnh[2 * tid]     = (v0 - mean) * rstd * prg[2 * tid]     + prb[2 * tid];
        lnh[2 * tid + 1] = (v1 - mean) * rstd * prg[2 * tid + 1] + prb[2 * tid + 1];
        __syncthreads();
        float pre = bp1[tid];
        for (int k = 0; k < 512; ++k) pre += lnh[k] * Wp1[k * 256 + tid];
        h1[tid] = 0.5f * pre * (1.f + erff(pre * 0.70710678f));
        __syncthreads();
        if (tid < 2) {
            float acc = bp2[tid];
            for (int c = 0; c < 256; ++c) acc += h1[c] * Wp2[c * 2 + tid];
            out[O_PROP + b * 2 + tid] = acc;
        }
        __syncthreads();
    }
}

// ---------------------------------------------------------------------------
extern "C" void kernel_launch(void* const* d_in, const int* in_sizes, int n_in,
                              void* d_out, int out_size, void* d_ws, size_t ws_size,
                              hipStream_t stream) {
    (void)in_sizes; (void)n_in; (void)out_size; (void)ws_size;
    FBParams P;
    for (int i = 0; i < 41; ++i) P.p[i] = (const float*)d_in[i];
    unsigned char* ws = (unsigned char*)d_ws;
    float* out = (float*)d_out;

    const int lds1 = 171264;   // recurrence LDS (per-WGP, dynamic)
    const int lds2 = 143616;   // thalamus LDS (double-buffered async A tile)

    (void)hipFuncSetAttribute((const void*)fb_recurrence,
                              hipFuncAttributeMaxDynamicSharedMemorySize, lds1);
    (void)hipFuncSetAttribute((const void*)fb_thalamus,
                              hipFuncAttributeMaxDynamicSharedMemorySize, lds2);

    fb_prep_small<<<1, 512, 0, stream>>>(P, ws);
    fb_prep_pack<<<1024, 256, 0, stream>>>(P, ws);
    fb_recurrence<<<2, 512, lds1, stream>>>(P, ws, out);
    fb_thalamus<<<2048, 512, lds2, stream>>>(P, ws, out);
    fb_heads<<<8192, 256, 0, stream>>>(P, out);
    fb_props<<<1, 256, 0, stream>>>(P, out);
}
